// KAN_2_40123584479542
// MI455X (gfx1250) — matrix operand order repaired and checked by reading, not compiled, not run on hardware
//
#include <hip/hip_runtime.h>
#include <hip/hip_bf16.h>

// ---------------------------------------------------------------------------
// KAN forward for MI455X (gfx1250), wave32 + WMMA f16->f32 + async LDS loads.
//
// Each KAN layer is one GEMM:  [B, I*9] x [I*9, O]
//   feature j==0  : silu(x_i)          (weight row = sb[i][o])
//   feature j=1..8: Bspline_j(x_i)     (weight row = coef[i][o][j-1]*ss[i][o])
// The f16 B panel is streamed into LDS with global_load_async_to_lds_b128
// (overlapped with the VALU spline-feature build), A tile staged in LDS,
// inner loop = ds_load_b128 pairs -> v_wmma_f32_16x16x32_f16.
// 256 threads = 8 waves -> 2 waves/SIMD32 on the WGP for latency hiding.
// ---------------------------------------------------------------------------

typedef __attribute__((ext_vector_type(16))) _Float16 v16h;
typedef __attribute__((ext_vector_type(8)))  _Float16 v8h;
typedef __attribute__((ext_vector_type(8)))  float    v8f;

#define BATCH 65536
#define NX 32
#define H  64
#define MID 129
#define KPA 576      // 64*9, multiple of 32
#define NPA 144      // pad(129) -> 9 N-tiles
#define KPB 1184     // pad(129*9=1161) -> 37 K-tiles
#define NPB 64

// --------------------------- feature builder -------------------------------
// Cox-de Boor cubic B-splines on uniform extended grid: GRID=5, K=3,
// knots g[p] = (p-3)*0.4 - 1, p = 0..11.  Produces silu + 8 bases.
__device__ __forceinline__ void kan_features(float x, _Float16* __restrict__ dst) {
    const float step = 0.4f;
    float sig = 1.0f / (1.0f + __expf(-x));
    dst[0] = (_Float16)(x * sig);                 // silu feature (j==0)

    float Bv[11];
#pragma unroll
    for (int p = 0; p < 11; ++p) {
        float g0 = (float)(p - 3) * step - 1.0f;
        float g1 = (float)(p - 2) * step - 1.0f;
        Bv[p] = (x >= g0 && x < g1) ? 1.0f : 0.0f;
    }
#pragma unroll
    for (int p = 1; p <= 3; ++p) {
        float inv = 1.0f / ((float)p * step);
#pragma unroll
        for (int m = 0; m < 10; ++m) {
            if (m + p < 11) {
                float gm  = (float)(m - 3) * step - 1.0f;          // g[m]
                float gm1 = (float)(m - 2) * step - 1.0f;          // g[m+1]
                float left  = (x - gm) * inv;                      // /(g[m+p]-g[m])
                float right = ((gm1 + (float)p * step) - x) * inv; // (g[m+p+1]-x)/...
                Bv[m] = left * Bv[m] + right * Bv[m + 1];
            }
        }
    }
#pragma unroll
    for (int j = 0; j < 8; ++j) dst[1 + j] = (_Float16)Bv[j];
}

// --------------------------- weight panel pack -----------------------------
// dst: f16, column-major [NP][KP]
__global__ __launch_bounds__(256) void prep_panel_kernel(
    const float* __restrict__ coef, const float* __restrict__ sb,
    const float* __restrict__ ss, int I, int O, int KP, int NP,
    _Float16* __restrict__ dst)
{
    int idx = blockIdx.x * 256 + threadIdx.x;
    if (idx >= NP * KP) return;
    int col = idx / KP;
    int k   = idx - col * KP;
    float v = 0.0f;
    if (col < O && k < I * 9) {
        int i  = k / 9;
        int j  = k - i * 9;
        int io = i * O + col;
        v = (j == 0) ? sb[io] : coef[io * 8 + (j - 1)] * ss[io];
    }
    dst[idx] = (_Float16)v;
}

// --------------------------- input layer (relu GEMM) -----------------------
__global__ __launch_bounds__(256) void input_layer_kernel(
    const float* __restrict__ x, const float* __restrict__ Win,
    const float* __restrict__ bin, float* __restrict__ h)
{
    int t = blockIdx.x * 256 + threadIdx.x;       // BATCH*H threads
    int row = t >> 6;
    int o   = t & 63;
    float acc = bin[o];
#pragma unroll
    for (int i = 0; i < NX; ++i)
        acc = fmaf(x[row * NX + i], Win[i * H + o], acc);
    h[t] = fmaxf(acc, 0.0f);
}

// --------------------------- fused KAN layer -------------------------------
// 256 threads = 8 waves.  MT = M tiles per block (8 -> 128 rows, 4 -> 64 rows).
// MT==8: each wave owns one 16-row M tile and all N tiles.
// MT==4: two waves share an M tile and interleave the N tiles.
__global__ __launch_bounds__(256) void kan_layer_kernel(
    const float* __restrict__ in, int I, int inStride,
    const _Float16* __restrict__ panel, int KP, int NT, int MT,
    const float* __restrict__ bias,
    float* __restrict__ out, int O, int outStride)
{
    extern __shared__ _Float16 smem[];
    const int NP    = NT * 16;
    const int MROWS = MT * 16;
    _Float16* Bs = smem;                            // [NP][KP] col-major, LDS off 0
    _Float16* As = smem + (size_t)NP * KP;          // [MROWS][KP] row-major

    const int tid     = threadIdx.x;
    const int rowBase = blockIdx.x * MROWS;

    // ---- async-stream B panel into LDS (ASYNCcnt path, no VGPR round trip) ----
    {
        const int chunks = (NP * KP) >> 3;          // 16-byte chunks; Bs at LDS 0
        for (int c = tid; c < chunks; c += 256) {
            unsigned ldsv = (unsigned)(c << 4);     // LDS byte address
            unsigned voff = (unsigned)(c << 4);     // byte offset from panel base
            asm volatile("global_load_async_to_lds_b128 %0, %1, %2"
                         :: "v"(ldsv), "v"(voff), "s"(panel) : "memory");
        }
    }

    // ---- overlap: build f16 feature tile (silu + 8 spline bases) ----
    {
        const int pairs = MROWS * I;
        for (int idx = tid; idx < pairs; idx += 256) {
            int r = idx / I;
            int i = idx - r * I;
            float xv = in[(size_t)(rowBase + r) * inStride + i];
            kan_features(xv, As + (size_t)r * KP + i * 9);
        }
        const int tail = KP - I * 9;                // zero-pad K (layer b)
        if (tail > 0) {
            for (int idx = tid; idx < MROWS * tail; idx += 256) {
                int r = idx / tail;
                int k = I * 9 + (idx - r * tail);
                As[(size_t)r * KP + k] = (_Float16)0.0f;
            }
        }
    }
    asm volatile("s_wait_asynccnt 0x0" ::: "memory");
    __syncthreads();

    // ---- WMMA main loop ----
    const int wave   = tid >> 5;
    const int lane   = tid & 31;
    const int laneLo = lane & 15;
    const int laneHi = lane >> 4;
    const int KT     = KP >> 5;

    const int wm     = wave & (MT - 1);             // M tile of this wave
    const int ntBase = wave / MT;                   // first N tile
    const int ntStep = 8 / MT;                      // N tile stride

    const _Float16* aRow = As + (size_t)(wm * 16 + laneLo) * KP;

    for (int nt = ntBase; nt < NT; nt += ntStep) {
        v8f acc = {};
        const _Float16* bCol = Bs + (size_t)(nt * 16 + laneLo) * KP;
        for (int kt = 0; kt < KT; ++kt) {
            // A 16x32 f16 fragment: lane<16 -> K{0..7,16..23}, lane>=16 -> +8
            int ka = (kt << 5) + laneHi * 8;
            v8h a0 = *(const v8h*)(aRow + ka);
            v8h a1 = *(const v8h*)(aRow + ka + 16);
            // B 32x16 f16 fragment: column laneLo, lane<16 -> K{0..7,16..23}, lane>=16 -> +8
            int kb = (kt << 5) + laneHi * 8;
            v8h b0 = *(const v8h*)(bCol + kb);
            v8h b1 = *(const v8h*)(bCol + kb + 16);
            v16h av = __builtin_shufflevector(a0, a1, 0,1,2,3,4,5,6,7,8,9,10,11,12,13,14,15);
            v16h bv = __builtin_shufflevector(b0, b1, 0,1,2,3,4,5,6,7,8,9,10,11,12,13,14,15);
            acc = __builtin_amdgcn_wmma_f32_16x16x32_f16(
                false, av, false, bv, (short)0, acc, false, false);
        }
        // D 16x16 f32: VGPR v -> row v (lanes 0-15) / v+8 (lanes 16-31)
        int col = nt * 16 + laneLo;
        if (col < O) {
            float bv = bias[col];
            int r0 = rowBase + wm * 16 + laneHi * 8;
#pragma unroll
            for (int v = 0; v < 8; ++v)
                out[(size_t)(r0 + v) * outStride + col] = acc[v] + bv;
        }
    }
}

// --------------------------- output head (sigmoid) -------------------------
__global__ __launch_bounds__(256) void output_layer_kernel(
    const float* __restrict__ h, const float* __restrict__ Wout,
    float* __restrict__ y)
{
    int row = blockIdx.x * 256 + threadIdx.x;
    float acc = 0.0f;
#pragma unroll
    for (int i = 0; i < H; ++i)
        acc = fmaf(h[row * H + i], Wout[i], acc);
    y[row] = 1.0f / (1.0f + __expf(-acc));
}

// ---------------------------------------------------------------------------
extern "C" void kernel_launch(void* const* d_in, const int* in_sizes, int n_in,
                              void* d_out, int out_size, void* d_ws, size_t ws_size,
                              hipStream_t stream)
{
    (void)in_sizes; (void)n_in; (void)out_size; (void)ws_size;

    const float* x      = (const float*)d_in[0];
    const float* Win    = (const float*)d_in[1];
    const float* bin    = (const float*)d_in[2];
    const float* Wout   = (const float*)d_in[3];
    const float* coef0a = (const float*)d_in[4];
    const float* sb0a   = (const float*)d_in[5];
    const float* ss0a   = (const float*)d_in[6];
    const float* bias0a = (const float*)d_in[7];
    const float* coef0b = (const float*)d_in[8];
    const float* sb0b   = (const float*)d_in[9];
    const float* ss0b   = (const float*)d_in[10];
    const float* bias0b = (const float*)d_in[11];
    const float* coef1a = (const float*)d_in[12];
    const float* sb1a   = (const float*)d_in[13];
    const float* ss1a   = (const float*)d_in[14];
    const float* bias1a = (const float*)d_in[15];
    const float* coef1b = (const float*)d_in[16];
    const float* sb1b   = (const float*)d_in[17];
    const float* ss1b   = (const float*)d_in[18];
    const float* bias1b = (const float*)d_in[19];

    // workspace layout
    char* ws = (char*)d_ws;
    float*     WH  = (float*)ws;                                   // [B][64]
    float*     WM  = (float*)(ws + (size_t)BATCH * H * 4);         // [B][129]
    _Float16*  P0a = (_Float16*)(ws + (size_t)BATCH * H * 4
                                    + (size_t)BATCH * MID * 4);
    _Float16*  P0b = P0a + (size_t)NPA * KPA;
    _Float16*  P1a = P0b + (size_t)NPB * KPB;
    _Float16*  P1b = P1a + (size_t)NPA * KPA;

    // pack f16 B panels
    const int na = NPA * KPA, nb = NPB * KPB;
    prep_panel_kernel<<<(na + 255) / 256, 256, 0, stream>>>(coef0a, sb0a, ss0a, H,   MID, KPA, NPA, P0a);
    prep_panel_kernel<<<(nb + 255) / 256, 256, 0, stream>>>(coef0b, sb0b, ss0b, MID, H,   KPB, NPB, P0b);
    prep_panel_kernel<<<(na + 255) / 256, 256, 0, stream>>>(coef1a, sb1a, ss1a, H,   MID, KPA, NPA, P1a);
    prep_panel_kernel<<<(nb + 255) / 256, 256, 0, stream>>>(coef1b, sb1b, ss1b, MID, H,   KPB, NPB, P1b);

    // h0 = relu(x @ Win + bin)
    input_layer_kernel<<<(BATCH * H) / 256, 256, 0, stream>>>(x, Win, bin, WH);

    // LDS: layer a = (144*576 + 128*576)*2 = 313,344 B;  layer b = 303,104 B
    const size_t ldsA = (size_t)(NPA * KPA + 128 * KPA) * sizeof(_Float16);
    const size_t ldsB = (size_t)(NPB * KPB +  64 * KPB) * sizeof(_Float16);

    kan_layer_kernel<<<BATCH / 128, 256, ldsA, stream>>>(WH, H,   H,   P0a, KPA, NPA / 16, 8, bias0a, WM, MID, MID);
    kan_layer_kernel<<<BATCH /  64, 256, ldsB, stream>>>(WM, MID, MID, P0b, KPB, NPB / 16, 4, bias0b, WH, H,   H);
    kan_layer_kernel<<<BATCH / 128, 256, ldsA, stream>>>(WH, H,   H,   P1a, KPA, NPA / 16, 8, bias1a, WM, MID, MID);
    kan_layer_kernel<<<BATCH /  64, 256, ldsB, stream>>>(WM, MID, MID, P1b, KPB, NPB / 16, 4, bias1b, WH, H,   H);

    // y = sigmoid(h @ Wout)
    output_layer_kernel<<<BATCH / 256, 256, 0, stream>>>(WH, Wout, (float*)d_out);
}
